// ContinualLearningNetwork_64716567216566
// MI455X (gfx1250) — matrix operand order repaired and checked by reading, not compiled, not run on hardware
//
#include <hip/hip_runtime.h>
#include <hip/hip_bf16.h>

typedef unsigned short u16;
typedef unsigned int   u32;
typedef unsigned long long u64;

typedef __attribute__((ext_vector_type(16))) __bf16 v16bf;
typedef __attribute__((ext_vector_type(8)))  float  v8f;
typedef __attribute__((ext_vector_type(4)))  int    v4i;

// ---------------- constants of this problem ----------------
#define BSZ   2048
#define INDIM 2048
#define L1    512
#define L2    256
#define FEAT  256
#define MEMN  100000
#define HID   2048
#define OUTD  1024
#define TOPK  2048

// ---------------- CDNA5 async global->LDS path (guarded) ----------------
#if __has_builtin(__builtin_amdgcn_global_load_async_to_lds_b128)
#define HAVE_ASYNC_LDS 1
#else
#define HAVE_ASYNC_LDS 0
#endif

typedef __attribute__((address_space(1))) v4i* gvec_t;
typedef __attribute__((address_space(3))) v4i* lvec_t;

__device__ __forceinline__ void cp_async16(const void* g, void* l) {
#if HAVE_ASYNC_LDS
    __builtin_amdgcn_global_load_async_to_lds_b128((gvec_t)g, (lvec_t)l, 0, 0);
#else
    *reinterpret_cast<uint4*>(l) = *reinterpret_cast<const uint4*>(g);
#endif
}

__device__ __forceinline__ void wait_async0() {
#if HAVE_ASYNC_LDS
#if __has_builtin(__builtin_amdgcn_s_wait_asynccnt)
    __builtin_amdgcn_s_wait_asynccnt(0);
#else
    asm volatile("s_wait_asynccnt 0x0" ::: "memory");
#endif
#endif
}

// ---------------- helpers ----------------
__device__ __forceinline__ u16 f2bf(float x) {
    u32 u = __float_as_uint(x);
    u32 r = u + 0x7FFFu + ((u >> 16) & 1u);   // round-to-nearest-even
    return (u16)(r >> 16);
}

__device__ __forceinline__ u32 orderKey(float f) {
    u32 u = __float_as_uint(f);
    return (u & 0x80000000u) ? ~u : (u | 0x80000000u);   // monotonic float->u32
}

// ---------------- f32 -> bf16 bulk convert (n % 4 == 0) ----------------
__global__ void cvt_bf16_kernel(const float* __restrict__ in, u16* __restrict__ out, int n) {
    int i = (blockIdx.x * blockDim.x + threadIdx.x) * 4;
    if (i < n) {
        float4 v = *reinterpret_cast<const float4*>(in + i);
        out[i + 0] = f2bf(v.x);
        out[i + 1] = f2bf(v.y);
        out[i + 2] = f2bf(v.z);
        out[i + 3] = f2bf(v.w);
    }
}

// -------- fused f32->bf16 convert + transpose: W[K][N] -> WT[N][K] --------
__global__ __launch_bounds__(256) void cvt_transpose_kernel(
    const float* __restrict__ W, u16* __restrict__ WT, int K, int N)
{
    __shared__ u16 t[32][33];
    int n0 = blockIdx.x * 32;
    int k0 = blockIdx.y * 32;
    int tx = threadIdx.x & 31;
    int ty = threadIdx.x >> 5;                 // 0..7
#pragma unroll
    for (int r = ty; r < 32; r += 8)
        t[r][tx] = f2bf(W[(size_t)(k0 + r) * N + n0 + tx]);
    __syncthreads();
#pragma unroll
    for (int r = ty; r < 32; r += 8)
        WT[(size_t)(n0 + r) * K + k0 + tx] = t[tx][r];
}

// ---------------- bf16 WMMA GEMM ----------------
// C[M,N] = A[M,K] * BT[N,K]^T + bias, optional relu.
// 256 threads (8 wave32s), tile 128x128, K-step 64, double-buffered async LDS.
union FragAB { uint4 q[2]; v16bf v; };

#define KSTEP 64
#define SRU   72          // LDS row stride in u16 (64 data + 8 pad)
#define SRB   144         // LDS row stride in bytes

__global__ __launch_bounds__(256) void gemm_bf16_kernel(
    const u16* __restrict__ A, const u16* __restrict__ BT,
    const float* __restrict__ bias,
    float* __restrict__ outF, u16* __restrict__ outB,
    int M, int N, int K, int doRelu)
{
    __shared__ u16 sA[2][128 * SRU];
    __shared__ u16 sB[2][128 * SRU];

    const int tn0  = blockIdx.x * 128;
    const int tm0  = blockIdx.y * 128;
    const int tid  = threadIdx.x;
    const int lane = tid & 31;
    const int wv   = tid >> 5;
    const int mb   = (wv & 3) * 32;   // wave M offset in tile
    const int nb   = (wv >> 2) * 64;  // wave N offset in tile
    const int half = lane >> 4;
    const int l15  = lane & 15;

    const char* Ag = reinterpret_cast<const char*>(A);
    const char* Bg = reinterpret_cast<const char*>(BT);

    v8f acc[2][4];
    const v8f zero = {0.f, 0.f, 0.f, 0.f, 0.f, 0.f, 0.f, 0.f};
#pragma unroll
    for (int mi = 0; mi < 2; ++mi)
#pragma unroll
        for (int ni = 0; ni < 4; ++ni) acc[mi][ni] = zero;

    auto stage = [&](int buf, int kt) {
        char* la = reinterpret_cast<char*>(&sA[buf][0]);
        char* lb = reinterpret_cast<char*>(&sB[buf][0]);
#pragma unroll
        for (int i = 0; i < 4; ++i) {
            int c   = tid + 256 * i;        // 0..1023
            int row = c >> 3;
            int cb  = (c & 7) * 16;         // byte column within 128B row
            cp_async16(Ag + ((size_t)(tm0 + row) * K + kt) * 2 + cb, la + row * SRB + cb);
            cp_async16(Bg + ((size_t)(tn0 + row) * K + kt) * 2 + cb, lb + row * SRB + cb);
        }
    };

    stage(0, 0);                              // prologue
    int cur = 0;
    for (int kt = 0; kt < K; kt += KSTEP) {
        wait_async0();                        // my writes to buf[cur] complete
        __syncthreads();                      // all writes done; all prior reads done
        if (kt + KSTEP < K) stage(cur ^ 1, kt + KSTEP);

        const u16* la = &sA[cur][0];
        const u16* lb = &sB[cur][0];
#pragma unroll
        for (int s = 0; s < 2; ++s) {         // two 16x16x32 K-substeps
            const int kc = 32 * s;
            FragAB a[2], b[4];
#pragma unroll
            for (int mi = 0; mi < 2; ++mi) {
                int row = mb + 16 * mi + l15;
                a[mi].q[0] = *reinterpret_cast<const uint4*>(&la[row * SRU + kc + 8 * half]);
                a[mi].q[1] = *reinterpret_cast<const uint4*>(&la[row * SRU + kc + 16 + 8 * half]);
            }
#pragma unroll
            for (int ni = 0; ni < 4; ++ni) {
                int row = nb + 16 * ni + l15;
                b[ni].q[0] = *reinterpret_cast<const uint4*>(&lb[row * SRU + kc + 8 * half]);
                b[ni].q[1] = *reinterpret_cast<const uint4*>(&lb[row * SRU + kc + 16 + 8 * half]);
            }
#pragma unroll
            for (int mi = 0; mi < 2; ++mi)
#pragma unroll
                for (int ni = 0; ni < 4; ++ni)
                    acc[mi][ni] = __builtin_amdgcn_wmma_f32_16x16x32_bf16(
                        false, a[mi].v, false, b[ni].v, (short)0, acc[mi][ni], false, false);
        }
        cur ^= 1;
    }

    // ---- epilogue: bias + relu, f32 and/or bf16 stores ----
#pragma unroll
    for (int mi = 0; mi < 2; ++mi)
#pragma unroll
        for (int ni = 0; ni < 4; ++ni) {
            int col = tn0 + nb + 16 * ni + l15;
            float bv = bias[col];
#pragma unroll
            for (int v = 0; v < 8; ++v) {
                int row = tm0 + mb + 16 * mi + half * 8 + v;
                float x = acc[mi][ni][v] + bv;
                if (doRelu) x = fmaxf(x, 0.f);
                size_t o = (size_t)row * N + col;
                if (outF) outF[o] = x;
                if (outB) outB[o] = f2bf(x);
            }
        }
}

// ---------------- column sums / sum-of-squares of f [2048 x 256] ----------------
__global__ void colstats_kernel(const float* __restrict__ f,
                                float* __restrict__ fsum, float* __restrict__ fsqp) {
    __shared__ float s1[256], s2[256];
    int d = blockIdx.x, t = threadIdx.x;
    float a = 0.f, b = 0.f;
    for (int r = t; r < BSZ; r += 256) {
        float v = f[(size_t)r * FEAT + d];
        a += v; b += v * v;
    }
    s1[t] = a; s2[t] = b;
    __syncthreads();
    for (int s = 128; s > 0; s >>= 1) {
        if (t < s) { s1[t] += s1[t + s]; s2[t] += s2[t + s]; }
        __syncthreads();
    }
    if (t == 0) { fsum[d] = s1[0]; fsqp[d] = s2[0]; }
}

__global__ void fsq_reduce_kernel(const float* __restrict__ part, float* __restrict__ fsq) {
    __shared__ float s[256];
    int t = threadIdx.x;
    s[t] = part[t];
    __syncthreads();
    for (int st = 128; st > 0; st >>= 1) {
        if (t < st) s[t] += s[t + st];
        __syncthreads();
    }
    if (t == 0) fsq[0] = s[0];
}

// ---------------- scores over memory table; emit order keys --------------------
__global__ void scores_kernel(const float* __restrict__ mem,
                              const float* __restrict__ fsum,
                              const float* __restrict__ fsq,
                              u32* __restrict__ keys) {
    __shared__ float sF[FEAT];
    sF[threadIdx.x] = fsum[threadIdx.x];
    __syncthreads();
    int wv = threadIdx.x >> 5, lane = threadIdx.x & 31;
    int row = blockIdx.x * 8 + wv;
    const float* m = mem + (size_t)row * FEAT + lane * 8;
    float dot = 0.f, sq = 0.f;
#pragma unroll
    for (int j = 0; j < 8; ++j) {
        float v = m[j];
        dot += v * sF[lane * 8 + j];
        sq  += v * v;
    }
    for (int off = 16; off; off >>= 1) {
        dot += __shfl_xor(dot, off);
        sq  += __shfl_xor(sq,  off);
    }
    if (lane == 0) {
        float s = -((float)BSZ * sq - 2.0f * dot + fsq[0]);
        keys[row] = orderKey(s);
    }
}

// ---------------- top-k machinery ----------------
// meta: [0..2047]=hist, [2048]=ctrA, [2049]=ctrB, [2050]=selBucket,
//       [2051]=countAbove, [2052]=need
__global__ void zero_u32_kernel(u32* p, int n) {
    int i = blockIdx.x * blockDim.x + threadIdx.x;
    if (i < n) p[i] = 0u;
}

__global__ void hist_kernel(const u32* __restrict__ keys, u32* __restrict__ meta) {
    int i = blockIdx.x * blockDim.x + threadIdx.x;
    if (i < MEMN) atomicAdd(&meta[keys[i] >> 21], 1u);
}

__global__ void scan_kernel(u32* meta) {
    if (threadIdx.x == 0) {
        u32 cum = 0; int b = 2047;
        for (; b >= 0; --b) {
            u32 c = meta[b];
            if (cum + c >= (u32)TOPK) break;
            cum += c;
        }
        meta[2050] = (u32)b;
        meta[2051] = cum;
        meta[2052] = (u32)TOPK - cum;
    }
}

__global__ void compact_kernel(const u32* __restrict__ keys, u32* meta, u64* __restrict__ sel) {
    int i = blockIdx.x * blockDim.x + threadIdx.x;
    if (i >= MEMN) return;
    u32 key = keys[i];
    u32 b = key >> 21;
    u32 selB = meta[2050], cntA = meta[2051], need = meta[2052];
    u64 rec = ((u64)key << 32) | (u32)(~(u32)i);
    if (b > selB) {
        u32 p = atomicAdd(&meta[2048], 1u);
        sel[p] = rec;
    } else if (b == selB) {
        u32 p = atomicAdd(&meta[2049], 1u);
        if (p < need) sel[cntA + p] = rec;
    }
}

__global__ __launch_bounds__(1024) void sort_topk_kernel(const u64* __restrict__ sel,
                                                         u32* __restrict__ idx) {
    __shared__ u64 s[TOPK];
    for (int i = threadIdx.x; i < TOPK; i += 1024) s[i] = sel[i];
    __syncthreads();
    for (u32 k = 2; k <= (u32)TOPK; k <<= 1) {
        for (u32 j = k >> 1; j > 0; j >>= 1) {
            for (u32 t = threadIdx.x; t < (u32)TOPK; t += 1024) {
                u32 ixj = t ^ j;
                if (ixj > t) {
                    u64 x = s[t], y = s[ixj];
                    bool desc = ((t & k) == 0);
                    if (desc ? (x < y) : (x > y)) { s[t] = y; s[ixj] = x; }
                }
            }
            __syncthreads();
        }
    }
    for (int i = threadIdx.x; i < TOPK; i += 1024) idx[i] = ~((u32)s[i]);
}

// ---------------- gather memory rows + concat with f (bf16) --------------------
__global__ void gather_concat_kernel(const u16* __restrict__ fb,
                                     const float* __restrict__ mem,
                                     const u32* __restrict__ idx,
                                     u16* __restrict__ h) {
    int j = blockIdx.x, t = threadIdx.x;                 // 256 threads
    h[(size_t)j * (2 * FEAT) + t] = fb[(size_t)j * FEAT + t];
    u32 r = idx[j];
    h[(size_t)j * (2 * FEAT) + FEAT + t] = f2bf(mem[(size_t)r * FEAT + t]);
}

// =========================== host-side launcher ================================
extern "C" void kernel_launch(void* const* d_in, const int* in_sizes, int n_in,
                              void* d_out, int out_size, void* d_ws, size_t ws_size,
                              hipStream_t stream) {
    (void)in_sizes; (void)n_in; (void)out_size; (void)ws_size;

    const float* x    = (const float*)d_in[0];
    const float* W1   = (const float*)d_in[1];
    const float* b1   = (const float*)d_in[2];
    const float* W2   = (const float*)d_in[3];
    const float* b2   = (const float*)d_in[4];
    const float* W3   = (const float*)d_in[5];
    const float* b3   = (const float*)d_in[6];
    const float* mem  = (const float*)d_in[7];
    const float* Wt1  = (const float*)d_in[8];
    const float* bt1  = (const float*)d_in[9];
    const float* Wt2  = (const float*)d_in[10];
    const float* bt2  = (const float*)d_in[11];
    const float* Wo   = (const float*)d_in[12];
    const float* bo   = (const float*)d_in[13];
    float* out = (float*)d_out;

    // ---- workspace bump allocator ----
    char* base = (char*)d_ws;
    size_t off = 0;
    auto alloc = [&](size_t bytes) -> void* {
        void* r = base + off;
        off = (off + bytes + 255) & ~(size_t)255;
        return r;
    };

    u16* xb    = (u16*)alloc((size_t)BSZ * INDIM * 2);
    u16* W1t   = (u16*)alloc((size_t)INDIM * L1 * 2);         // [N][K] transposed bf16
    u16* W2t   = (u16*)alloc((size_t)L1 * L2 * 2);
    u16* W3t   = (u16*)alloc((size_t)L2 * FEAT * 2);
    u16* Wt1t  = (u16*)alloc((size_t)(2 * FEAT) * HID * 2);
    u16* Wt2t  = (u16*)alloc((size_t)HID * HID * 2);
    u16* Wot   = (u16*)alloc((size_t)HID * OUTD * 2);
    u16* f1b   = (u16*)alloc((size_t)BSZ * L1 * 2);
    u16* f2b   = (u16*)alloc((size_t)BSZ * L2 * 2);
    float* fF  = (float*)alloc((size_t)BSZ * FEAT * 4);
    u16* fb    = (u16*)alloc((size_t)BSZ * FEAT * 2);
    float* fsum = (float*)alloc(FEAT * 4);
    float* fsqp = (float*)alloc(FEAT * 4);
    float* fsq  = (float*)alloc(4);
    u32* keys   = (u32*)alloc((size_t)MEMN * 4);
    u32* meta   = (u32*)alloc(2064 * 4);
    u64* sel    = (u64*)alloc((size_t)TOPK * 8);
    u32* idx    = (u32*)alloc((size_t)TOPK * 4);
    u16* hb    = (u16*)alloc((size_t)BSZ * (2 * FEAT) * 2);
    u16* h1b   = (u16*)alloc((size_t)BSZ * HID * 2);
    u16* h2b   = (u16*)alloc((size_t)BSZ * HID * 2);

    // ---- stage activations (row-major) and weights (transposed) as bf16 ----
    {
        size_t n = (size_t)BSZ * INDIM;
        cvt_bf16_kernel<<<(int)((n / 4 + 255) / 256), 256, 0, stream>>>(x, xb, (int)n);
    }
    auto cvtT = [&](const float* W, u16* WT, int K, int N) {
        cvt_transpose_kernel<<<dim3(N / 32, K / 32), 256, 0, stream>>>(W, WT, K, N);
    };
    cvtT(W1,  W1t,  INDIM,    L1);
    cvtT(W2,  W2t,  L1,       L2);
    cvtT(W3,  W3t,  L2,       FEAT);
    cvtT(Wt1, Wt1t, 2 * FEAT, HID);
    cvtT(Wt2, Wt2t, HID,      HID);
    cvtT(Wo,  Wot,  HID,      OUTD);

    // ---- feature extractor ----
    gemm_bf16_kernel<<<dim3(L1 / 128, BSZ / 128), 256, 0, stream>>>(
        xb, W1t, b1, nullptr, f1b, BSZ, L1, INDIM, 1);
    gemm_bf16_kernel<<<dim3(L2 / 128, BSZ / 128), 256, 0, stream>>>(
        f1b, W2t, b2, nullptr, f2b, BSZ, L2, L1, 1);
    gemm_bf16_kernel<<<dim3(FEAT / 128, BSZ / 128), 256, 0, stream>>>(
        f2b, W3t, b3, fF, fb, BSZ, FEAT, L2, 0);

    // ---- memory scoring ----
    colstats_kernel<<<FEAT, 256, 0, stream>>>(fF, fsum, fsqp);
    fsq_reduce_kernel<<<1, 256, 0, stream>>>(fsqp, fsq);
    scores_kernel<<<MEMN / 8, 256, 0, stream>>>(mem, fsum, fsq, keys);

    // ---- top-2048 radix select + bitonic sort ----
    zero_u32_kernel<<<(2064 + 255) / 256, 256, 0, stream>>>(meta, 2064);
    hist_kernel<<<(MEMN + 255) / 256, 256, 0, stream>>>(keys, meta);
    scan_kernel<<<1, 32, 0, stream>>>(meta);
    compact_kernel<<<(MEMN + 255) / 256, 256, 0, stream>>>(keys, meta, sel);
    sort_topk_kernel<<<1, 1024, 0, stream>>>(sel, idx);

    // ---- gather + concat ----
    gather_concat_kernel<<<BSZ, 256, 0, stream>>>(fb, mem, idx, hb);

    // ---- task column ----
    gemm_bf16_kernel<<<dim3(HID / 128, BSZ / 128), 256, 0, stream>>>(
        hb, Wt1t, bt1, nullptr, h1b, BSZ, HID, 2 * FEAT, 1);
    gemm_bf16_kernel<<<dim3(HID / 128, BSZ / 128), 256, 0, stream>>>(
        h1b, Wt2t, bt2, nullptr, h2b, BSZ, HID, HID, 1);
    gemm_bf16_kernel<<<dim3(OUTD / 128, BSZ / 128), 256, 0, stream>>>(
        h2b, Wot, bo, out, nullptr, BSZ, OUTD, HID, 0);
}
